// MLPDeepSet_48412871360693
// MI455X (gfx1250) — matrix operand order repaired and checked by reading, not compiled, not run on hardware
//
#include <hip/hip_runtime.h>
#include <hip/hip_bf16.h>

typedef __attribute__((ext_vector_type(16))) __bf16 v16bf;
typedef __attribute__((ext_vector_type(8)))  __bf16 bfx8;
typedef __attribute__((ext_vector_type(8)))  float  v8f;

#define N_ATOMS   8192
#define N_EDGES   262144
#define DIM       256
#define KPAD0     544      // 533 padded to 17*32
#define KC0       17
#define KC1       8
#define SA        552      // LDS row stride (elems) for 544-col region (1104B, 16B aligned, bank-spread)
#define SB        264      // LDS row stride (elems) for 256-col region (528B)
#define WPB       2        // waves per block

#define PACK_W0_ELEMS            (16 * KC0 * 512)              // 139264
#define PACK_WS_ELEMS_PER_LAYER  (16 * KC1 * 512)              // 65536
#define PACK_WS_ELEMS            (6 * PACK_WS_ELEMS_PER_LAYER) // 393216

// ---------------------------------------------------------------------------
// Prep: pack weights into wave32 WMMA B-fragment layout (bf16) and init out.
// B layout (16-bit, 32x16): lane l holds column n=l&15; its 16 values are
// k = kc*32 + (l>=16 ? 8 : 0) + (j&7) + (j>=8 ? 16 : 0), j=0..15.
// ---------------------------------------------------------------------------
__global__ void prep_kernel(const float* __restrict__ W_in,
                            const float* __restrict__ Ws,
                            const float* __restrict__ b_out,
                            __bf16* __restrict__ packW0,
                            __bf16* __restrict__ packWs,
                            float* __restrict__ out)
{
    const int total = PACK_W0_ELEMS + PACK_WS_ELEMS + N_ATOMS;
    for (int idx = blockIdx.x * blockDim.x + threadIdx.x; idx < total;
         idx += gridDim.x * blockDim.x) {
        if (idx < PACK_W0_ELEMS) {
            int nt   = idx / (KC0 * 512);
            int rem  = idx % (KC0 * 512);
            int kc   = rem / 512;
            int li   = rem % 512;
            int lane = li >> 4, j = li & 15;
            int n = nt * 16 + (lane & 15);
            int k = kc * 32 + ((lane & 16) ? 8 : 0) + (j & 7) + ((j & 8) ? 16 : 0);
            float v = (k < 533) ? W_in[k * 256 + n] : 0.0f;
            packW0[idx] = (__bf16)v;
        } else if (idx < PACK_W0_ELEMS + PACK_WS_ELEMS) {
            int t    = idx - PACK_W0_ELEMS;
            int l    = t / PACK_WS_ELEMS_PER_LAYER;
            int rem  = t % PACK_WS_ELEMS_PER_LAYER;
            int nt   = rem / (KC1 * 512);
            int rem2 = rem % (KC1 * 512);
            int kc   = rem2 / 512;
            int li   = rem2 % 512;
            int lane = li >> 4, j = li & 15;
            int n = nt * 16 + (lane & 15);
            int k = kc * 32 + ((lane & 16) ? 8 : 0) + (j & 7) + ((j & 8) ? 16 : 0);
            packWs[t] = (__bf16)Ws[(l * 256 + k) * 256 + n];
        } else {
            out[idx - PACK_W0_ELEMS - PACK_WS_ELEMS] = b_out[0];
        }
    }
}

// ---------------------------------------------------------------------------
// A-fragment (16-bit, 16x32): lane l holds row m=l&15; element j has
// K = j + 8*(j>=8) + 8*(l>=16)  => two contiguous 8-elem runs at
// kc*32 + 8*(l>=16) and kc*32 + 16 + 8*(l>=16).
// ---------------------------------------------------------------------------
__device__ __forceinline__ v16bf load_afrag(const __bf16* __restrict__ arow, int kc)
{
    bfx8 alo = *(const bfx8*)(arow + kc * 32);
    bfx8 ahi = *(const bfx8*)(arow + kc * 32 + 16);
    return __builtin_shufflevector(alo, ahi, 0, 1, 2, 3, 4, 5, 6, 7,
                                   8, 9, 10, 11, 12, 13, 14, 15);
}

// 4 N-tiles x KC k-chunks with software-pipelined (double-buffered) B frags:
// the 4 B fragments for kc+1 are in flight while the 4 WMMAs for kc issue.
template <int KC>
__device__ __forceinline__ void gemm_group(const __bf16* __restrict__ arow,
                                           const __bf16* __restrict__ wp,
                                           v8f acc[4])
{
    v16bf bcur[4];
#pragma unroll
    for (int u = 0; u < 4; ++u)
        bcur[u] = *(const v16bf*)(wp + (u * KC) * 512);
#pragma unroll
    for (int kc = 0; kc < KC; ++kc) {
        v16bf bnxt[4];
        if (kc + 1 < KC) {
#pragma unroll
            for (int u = 0; u < 4; ++u)
                bnxt[u] = *(const v16bf*)(wp + (u * KC + kc + 1) * 512);
        }
        v16bf a = load_afrag(arow, kc);
#pragma unroll
        for (int u = 0; u < 4; ++u)
            acc[u] = __builtin_amdgcn_wmma_f32_16x16x32_bf16(
                false, a, false, bcur[u], (short)0, acc[u], false, false);
        if (kc + 1 < KC) {
#pragma unroll
            for (int u = 0; u < 4; ++u) bcur[u] = bnxt[u];
        }
    }
}

// One full 256-wide layer for a 16-edge tile: 4 groups of 4 N-tiles.
// C layout: VGPR r, lane l -> M = r + 8*(l>=16), N = nt*16 + (l&15).
template <int KC, bool SILU>
__device__ __forceinline__ void layer4(const __bf16* __restrict__ src, int strideS,
                                       __bf16* __restrict__ dst, int strideD,
                                       const __bf16* __restrict__ wpack,
                                       const float* __restrict__ bias, int lane)
{
    const int nlo = lane & 15;
    const int mhi = (lane >> 4) * 8;
    const int c0  = (lane & 16) ? 8 : 0;
    const __bf16* arow = src + (lane & 15) * strideS + c0;

    for (int g = 0; g < 4; ++g) {
        v8f acc[4] = {};                      // zero-init -> inline-0 SRC2 on first WMMA
        const __bf16* wp = wpack + (g * 4) * KC * 512 + lane * 16;
        gemm_group<KC>(arow, wp, acc);
#pragma unroll
        for (int u = 0; u < 4; ++u) {
            const int nt = g * 4 + u;
            const float bval = bias[nt * 16 + nlo];
#pragma unroll
            for (int r = 0; r < 8; ++r) {
                float h = acc[u][r] + bval;
                if (SILU) h = h / (1.0f + __expf(-h));   // silu(x) = x*sigmoid(x)
                dst[(r + mhi) * strideD + nt * 16 + nlo] = (__bf16)h;
            }
        }
    }
    asm volatile("s_wait_dscnt 0x0" ::: "memory");
}

// ---------------------------------------------------------------------------
// Fused pipeline: gather pair features -> 7 GEMM layers -> per-edge dot with
// W_out -> atomic scatter-add onto atoms. One wave per 16-edge tile.
// ---------------------------------------------------------------------------
__global__ __launch_bounds__(32 * WPB) void fused_kernel(
    const float* __restrict__ x, const float* __restrict__ d_ij,
    const int* __restrict__ idx_i, const int* __restrict__ idx_j,
    const float* __restrict__ b_in, const float* __restrict__ bs,
    const float* __restrict__ W_out,
    const __bf16* __restrict__ packW0, const __bf16* __restrict__ packWs,
    float* __restrict__ out)
{
    __shared__ __bf16 lds[WPB][16 * SA + 16 * SB];
    const int wave = threadIdx.x >> 5;
    const int lane = threadIdx.x & 31;
    __bf16* A = &lds[wave][0];
    __bf16* B = &lds[wave][16 * SA];
    const int tile = blockIdx.x * WPB + wave;
    const int e0   = tile * 16;

    // ---- phase 1: build pair matrix [16 x 544] bf16 in region A ----
    for (int m = 0; m < 16; ++m) {
        const int ii = idx_i[e0 + m];
        const int jj = idx_j[e0 + m];
        const float* xi = x + (size_t)ii * DIM;
        const float* xj = x + (size_t)jj * DIM;
        for (int c = lane; c < DIM; c += 32) {
            A[m * SA + c]       = (__bf16)xi[c];
            A[m * SA + DIM + c] = (__bf16)xj[c];
        }
    }
    if (lane < 16) {
        const float width = 15.0f / 19.0f;          // linspace(0,15,20) spacing
        const float coeff = -0.5f / (width * width);
        float d = d_ij[e0 + lane];
#pragma unroll
        for (int c = 0; c < 20; ++c) {
            float t = d - width * (float)c;
            A[lane * SA + 2 * DIM + c] = (__bf16)__expf(coeff * t * t);
        }
        A[lane * SA + 532] = (__bf16)d;
#pragma unroll
        for (int c = 533; c < KPAD0; ++c) A[lane * SA + c] = (__bf16)0.0f;
    }
    asm volatile("s_wait_dscnt 0x0" ::: "memory");

    // ---- phase 2: layers (ping-pong A <-> B) ----
    const int PW = PACK_WS_ELEMS_PER_LAYER;
    layer4<KC0, false>(A, SA, B, SB, packW0,          b_in,       lane); // pair @ W_in
    layer4<KC1, true >(B, SB, A, SA, packWs + 0 * PW, bs + 0*256, lane); // Ws[0] + silu
    layer4<KC1, true >(A, SA, B, SB, packWs + 1 * PW, bs + 1*256, lane);
    layer4<KC1, true >(B, SB, A, SA, packWs + 2 * PW, bs + 2*256, lane);
    layer4<KC1, true >(A, SA, B, SB, packWs + 3 * PW, bs + 3*256, lane);
    layer4<KC1, true >(B, SB, A, SA, packWs + 4 * PW, bs + 4*256, lane);

    // ---- phase 3: last layer (Ws[5], no silu) with W_out folded in ----
    float part[8] = {0, 0, 0, 0, 0, 0, 0, 0};
    const int nlo = lane & 15;
    const int c0  = (lane & 16) ? 8 : 0;
    const __bf16* arow = A + (lane & 15) * SA + c0;
    for (int g = 0; g < 4; ++g) {
        v8f acc[4] = {};
        const __bf16* wp = packWs + 5 * PW + (g * 4) * KC1 * 512 + lane * 16;
        gemm_group<KC1>(arow, wp, acc);
#pragma unroll
        for (int u = 0; u < 4; ++u) {
            const int nt = g * 4 + u;
            const float bval = bs[5 * 256 + nt * 16 + nlo];
            const float w    = W_out[nt * 16 + nlo];
#pragma unroll
            for (int r = 0; r < 8; ++r)
                part[r] = __builtin_fmaf(acc[u][r] + bval, w, part[r]);
        }
    }

    // cross-lane reduce: part[r] on lane l covers (M = r + 8*(l>=16), N = l&15)
    float* scr = (float*)B;  // region B free now
#pragma unroll
    for (int r = 0; r < 8; ++r) scr[lane * 8 + r] = part[r];
    asm volatile("s_wait_dscnt 0x0" ::: "memory");
    if (lane < 16) {
        const int r    = lane & 7;
        const int base = (lane >> 3) * 16;
        float v = 0.0f;
#pragma unroll
        for (int j = 0; j < 16; ++j) v += scr[(base + j) * 8 + r];
        atomicAdd(&out[idx_i[e0 + lane]], v);
    }
}

// ---------------------------------------------------------------------------
extern "C" void kernel_launch(void* const* d_in, const int* in_sizes, int n_in,
                              void* d_out, int out_size, void* d_ws, size_t ws_size,
                              hipStream_t stream)
{
    const float* x     = (const float*)d_in[0];
    const float* dij   = (const float*)d_in[1];
    const int*   idx_i = (const int*)d_in[2];
    const int*   idx_j = (const int*)d_in[3];
    const float* W_in  = (const float*)d_in[4];
    const float* b_in  = (const float*)d_in[5];
    const float* Ws    = (const float*)d_in[6];
    const float* bs    = (const float*)d_in[7];
    const float* W_out = (const float*)d_in[8];
    const float* b_out = (const float*)d_in[9];
    float* out = (float*)d_out;

    __bf16* packW0 = (__bf16*)d_ws;
    __bf16* packWs = packW0 + PACK_W0_ELEMS;

    prep_kernel<<<2112, 256, 0, stream>>>(W_in, Ws, b_out, packW0, packWs, out);

    const int tiles = N_EDGES / 16;              // 16384 tiles, one per wave
    fused_kernel<<<tiles / WPB, 32 * WPB, 0, stream>>>(
        x, dij, idx_i, idx_j, b_in, bs, W_out, packW0, packWs, out);
}
